// GINO_37915971289741
// MI455X (gfx1250) — matrix operand order repaired and checked by reading, not compiled, not run on hardware
//
#include <hip/hip_runtime.h>
#include <hip/hip_bf16.h>
#include <math.h>

// Problem constants (from reference)
#define BB        2
#define NPTS      8192
#define WID_      64
#define GSZ       64
#define NG        4096        // G*G
#define KNN       16
#define M1_       12
#define M2_       12
#define MR_       24          // 2*M1 kept mode-rows
#define EDGE_IN_  71
#define EDGE_PAD  96

typedef __attribute__((ext_vector_type(16))) _Float16 v16h;
typedef __attribute__((ext_vector_type(8)))  _Float16 v8h;
typedef __attribute__((ext_vector_type(8)))  float    v8f;

__device__ __forceinline__ int lane_id() { return (int)(threadIdx.x & 31); }

__device__ __forceinline__ v8f v8f_zero() {
  v8f z;
#pragma unroll
  for (int i = 0; i < 8; ++i) z[i] = 0.0f;
  return z;
}

__device__ __forceinline__ float gelu_f(float x) {
  return 0.5f * x * (1.0f + erff(x * 0.70710678118654752f));
}

__device__ __forceinline__ float grid_ax(int i) {
  return -1.0f + (2.0f / 63.0f) * (float)i;   // jnp.linspace(-1,1,64)
}

// ---- WMMA operand handling (layouts per CDNA5 ISA 7.12.2) ----
// A fragment: 16x32 f16 tile from row-major LDS [16][ld] at column kbase.
// Per lane: halfs 0..7 = K[base..base+8), halfs 8..15 = K[base+16..base+24),
// base = kbase + (lane>=16 ? 8 : 0). Both runs are 16B aligned -> ds_load_b128.
__device__ __forceinline__ v16h load_A_rm(const _Float16* base, int ld, int kbase) {
  int lane = lane_id();
  const _Float16* p = base + (lane & 15) * ld + kbase + ((lane >> 4) << 3);
  v8h lo = *(const v8h*)p;
  v8h hi = *(const v8h*)(p + 16);
  return __builtin_shufflevector(lo, hi, 0, 1, 2, 3, 4, 5, 6, 7,
                                         8, 9, 10, 11, 12, 13, 14, 15);
}

// B fragments are pre-packed in LDS: tile (kt,nt) occupies 512 contiguous halfs;
// lane l's 16 halfs live at [tile*512 + l*16]. Lane l holds column n = nt*16+(l&15),
// rows K = kt*32 + (l>=16?8:0) + {0..7, 16..23}  (matches A fragment k order).
__device__ __forceinline__ void stage_B_packed(_Float16* dst, const float* W,
                                               int Ktiles, int Kvalid, int trans) {
  int total = Ktiles * 4 * 512;
  for (int p = (int)threadIdx.x; p < total; p += (int)blockDim.x) {
    int tile = p >> 9;
    int l = (p >> 4) & 31;
    int i = p & 15;
    int kt = tile >> 2, nt = tile & 3;
    int n = nt * 16 + (l & 15);
    int k = kt * 32 + ((l >> 4) << 3) + (i & 7) + ((i >> 3) << 4);
    float v = 0.0f;
    if (k < Kvalid) v = trans ? W[n * 64 + k] : W[k * 64 + n];
    dst[p] = (_Float16)v;
  }
}
__device__ __forceinline__ v16h load_B_packed(const _Float16* w, int kt, int nt) {
  int lane = lane_id();
  const _Float16* p = w + (((kt * 4 + nt) << 9) + (lane << 4));
  v8h lo = *(const v8h*)p;
  v8h hi = *(const v8h*)(p + 8);
  return __builtin_shufflevector(lo, hi, 0, 1, 2, 3, 4, 5, 6, 7,
                                         8, 9, 10, 11, 12, 13, 14, 15);
}

__device__ __forceinline__ v8f wmma32(v16h a, v16h b, v8f c) {
  return __builtin_amdgcn_wmma_f32_16x16x32_f16(false, a, false, b, (short)0, c, false, false);
}

// D tile (16x16 f32) -> f16 row-major [16][ld] at column nbase.
__device__ __forceinline__ void store_D_f16(v8f c, _Float16* out, int ld, int nbase) {
  int lane = lane_id();
  int n = nbase + (lane & 15), mo = (lane >> 4) * 8;
#pragma unroll
  for (int r = 0; r < 8; ++r) out[(mo + r) * ld + n] = (_Float16)c[r];
}

// ============================ lift MLP (6->64 GELU ->64) ============================
__global__ __launch_bounds__(256) void k_lift(const float* __restrict__ feats,
                                              const float* __restrict__ W1, const float* __restrict__ b1,
                                              const float* __restrict__ W2, const float* __restrict__ b2,
                                              float* __restrict__ out) {
  __shared__ alignas(16) _Float16 sW1[1 * 4 * 512];
  __shared__ alignas(16) _Float16 sW2[2 * 4 * 512];
  __shared__ alignas(16) _Float16 sA[8][16 * 64];
  int tid = threadIdx.x;
  stage_B_packed(sW1, W1, 1, 6, 0);
  stage_B_packed(sW2, W2, 2, 64, 0);
  __syncthreads();
  int wid = tid >> 5, lane = tid & 31;
  int row0 = (blockIdx.x * 8 + wid) * 16;
  _Float16* A = sA[wid];
  for (int t = lane; t < 16 * 32; t += 32) {
    int m = t >> 5, k = t & 31;
    A[m * 64 + k] = (_Float16)(k < 6 ? feats[(size_t)(row0 + m) * 6 + k] : 0.0f);
  }
  v8f acc[4];
#pragma unroll
  for (int nt = 0; nt < 4; ++nt) acc[nt] = v8f_zero();
  {
    v16h a = load_A_rm(A, 64, 0);
#pragma unroll
    for (int nt = 0; nt < 4; ++nt)
      acc[nt] = wmma32(a, load_B_packed(sW1, 0, nt), acc[nt]);
  }
#pragma unroll
  for (int nt = 0; nt < 4; ++nt) {
    int n = nt * 16 + (lane & 15);
    float bias = b1[n];
#pragma unroll
    for (int r = 0; r < 8; ++r) acc[nt][r] = gelu_f(acc[nt][r] + bias);
    store_D_f16(acc[nt], A, 64, nt * 16);
  }
  v8f acc2[4];
#pragma unroll
  for (int nt = 0; nt < 4; ++nt) acc2[nt] = v8f_zero();
#pragma unroll
  for (int kc = 0; kc < 2; ++kc) {
    v16h a = load_A_rm(A, 64, kc * 32);
#pragma unroll
    for (int nt = 0; nt < 4; ++nt)
      acc2[nt] = wmma32(a, load_B_packed(sW2, kc, nt), acc2[nt]);
  }
#pragma unroll
  for (int nt = 0; nt < 4; ++nt) {
    int n = nt * 16 + (lane & 15), mo = (lane >> 4) * 8;
    float bias = b2[n];
#pragma unroll
    for (int r = 0; r < 8; ++r) out[(size_t)(row0 + mo + r) * 64 + n] = acc2[nt][r] + bias;
  }
}

// ============================ brute-force KNN (one wave per query) ============================
__global__ __launch_bounds__(256) void k_topk(const float* __restrict__ qry, const float* __restrict__ src,
                                              int Nq, int Ns, int qry_grid, int src_grid,
                                              float* __restrict__ out_dist, int* __restrict__ out_idx) {
  __shared__ float md[8][512];
  __shared__ int   mi[8][512];
  int wid = threadIdx.x >> 5, lane = lane_id();
  int gq = blockIdx.x * 8 + wid;
  int b = gq / Nq, q = gq - b * Nq;
  float qx, qy;
  if (qry_grid) { qx = grid_ax(q >> 6); qy = grid_ax(q & 63); }
  else { const float* p = qry + (size_t)(b * Nq + q) * 2; qx = p[0]; qy = p[1]; }
  float dl[KNN]; int il[KNN];
#pragma unroll
  for (int j = 0; j < KNN; ++j) { dl[j] = 3.0e38f; il[j] = 0; }
  for (int s = lane; s < Ns; s += 32) {
    float sx, sy;
    if (src_grid) { sx = grid_ax(s >> 6); sy = grid_ax(s & 63); }
    else {
      const float* p = src + (size_t)(b * Ns + s) * 2;
      sx = p[0]; sy = p[1];
      if (s + 128 < Ns) __builtin_prefetch(src + (size_t)(b * Ns + s + 128) * 2, 0, 1);
    }
    float dx = qx - sx, dy = qy - sy;
    float d = sqrtf(dx * dx + dy * dy);
    if (d < dl[KNN - 1]) {
      float cd = d; int ci = s;
#pragma unroll
      for (int j = 0; j < KNN; ++j) {
        bool sw = cd < dl[j];
        float td = sw ? dl[j] : cd;
        int   ti = sw ? il[j] : ci;
        dl[j] = sw ? cd : dl[j];
        il[j] = sw ? ci : il[j];
        cd = td; ci = ti;
      }
    }
  }
#pragma unroll
  for (int j = 0; j < KNN; ++j) { md[wid][lane * KNN + j] = dl[j]; mi[wid][lane * KNN + j] = il[j]; }
  int ptr = 0;
  for (int k = 0; k < KNN; ++k) {
    float cand = (ptr < KNN) ? md[wid][lane * KNN + ptr] : 3.0e38f;
    int   cidx = (ptr < KNN) ? mi[wid][lane * KNN + ptr] : 0;
    float bd = cand; int bl = lane;
#pragma unroll
    for (int off = 16; off > 0; off >>= 1) {
      float od = __shfl_xor(bd, off, 32);
      int   ol = __shfl_xor(bl, off, 32);
      if (od < bd || (od == bd && ol < bl)) { bd = od; bl = ol; }
    }
    if (lane == bl) {
      out_dist[(size_t)gq * KNN + k] = cand;
      out_idx [(size_t)gq * KNN + k] = cidx;
      ++ptr;
    }
  }
}

// ============================ exact median of farthest-NN distance (radix select) ============================
__global__ __launch_bounds__(256) void k_median_sigma(const float* __restrict__ dist, int n,
                                                      float* __restrict__ sigma_out) {
  __shared__ unsigned hist[256];
  __shared__ unsigned s_newt, s_prefix, s_mask;
  int tid = threadIdx.x;
  if (tid == 0) { s_prefix = 0u; s_mask = 0u; s_newt = (unsigned)((n - 1) / 2); }
  __syncthreads();
  for (int shift = 24; shift >= 0; shift -= 8) {
    for (int i = tid; i < 256; i += blockDim.x) hist[i] = 0u;
    __syncthreads();
    unsigned prefix = s_prefix, mask = s_mask;
    for (int i = tid; i < n; i += blockDim.x) {
      unsigned u = __float_as_uint(dist[(size_t)i * KNN + (KNN - 1)]);  // dists >= 0: bits order-preserving
      if ((u & mask) == prefix) atomicAdd(&hist[(u >> shift) & 255u], 1u);
    }
    __syncthreads();
    if (tid == 0) {
      unsigned t = s_newt, c = 0u, sel = 255u;
      for (unsigned bn = 0; bn < 256; ++bn) {
        unsigned h = hist[bn];
        if (c + h > t) { sel = bn; s_newt = t - c; break; }
        c += h;
      }
      s_prefix = prefix | (sel << shift);
      s_mask = mask | (255u << shift);
    }
    __syncthreads();
  }
  if (tid == 0) sigma_out[0] = fmaxf(__uint_as_float(s_prefix), 1e-6f);
}

// ============================ GNO edge MLP + weighted aggregation (one wave per query) ============================
__global__ __launch_bounds__(256) void k_edge(const float* __restrict__ qryc, const float* __restrict__ srcc,
                                              const float* __restrict__ srcf,
                                              const int* __restrict__ nn_idx, const float* __restrict__ nn_dist,
                                              const float* __restrict__ sigma_p,
                                              const float* __restrict__ W1, const float* __restrict__ b1,
                                              const float* __restrict__ W2, const float* __restrict__ b2,
                                              const float* __restrict__ W3, const float* __restrict__ b3,
                                              int Nq, int Ns, int qry_grid, int src_grid,
                                              int srcf_cm, int out_cm, float* __restrict__ out) {
  __shared__ alignas(16) _Float16 sW1[3 * 4 * 512];
  __shared__ alignas(16) _Float16 sW2[2 * 4 * 512];
  __shared__ alignas(16) _Float16 sW3[2 * 4 * 512];
  __shared__ alignas(16) _Float16 sE[8][16 * EDGE_PAD];
  __shared__ float    sD[8][16];
  __shared__ int      sI[8][16];
  int tid = threadIdx.x;
  stage_B_packed(sW1, W1, 3, EDGE_IN_, 0);
  stage_B_packed(sW2, W2, 2, 64, 0);
  stage_B_packed(sW3, W3, 2, 64, 0);
  __syncthreads();
  int wid = tid >> 5, lane = tid & 31;
  int gq = blockIdx.x * 8 + wid;
  int b = gq / Nq, q = gq - b * Nq;
  float sigma = sigma_p[0];
  float qx, qy;
  if (qry_grid) { qx = grid_ax(q >> 6); qy = grid_ax(q & 63); }
  else { const float* p = qryc + (size_t)(b * Nq + q) * 2; qx = p[0]; qy = p[1]; }
  _Float16* E = sE[wid];
  if (lane < 16) {
    int k = lane;
    int idx = nn_idx[(size_t)gq * KNN + k];
    float d = nn_dist[(size_t)gq * KNN + k];
    float sx, sy;
    if (src_grid) { sx = grid_ax(idx >> 6); sy = grid_ax(idx & 63); }
    else { const float* p = srcc + (size_t)(b * Ns + idx) * 2; sx = p[0]; sy = p[1]; }
    _Float16* row = E + k * EDGE_PAD;
    row[0] = (_Float16)qx;        row[1] = (_Float16)qy;
    row[2] = (_Float16)sx;        row[3] = (_Float16)sy;
    row[4] = (_Float16)(qx - sx); row[5] = (_Float16)(qy - sy);
    row[6] = (_Float16)d;
    sD[wid][k] = d; sI[wid][k] = idx;
  }
  for (int t = lane; t < 16 * 64; t += 32) {      // neighbor features: cols 7..70
    int k = t >> 6, j = t & 63;
    int idx = sI[wid][k];
    float f = srcf_cm ? srcf[(size_t)(b * 64 + j) * 4096 + idx]
                      : srcf[(size_t)(b * Ns + idx) * 64 + j];
    E[k * EDGE_PAD + 7 + j] = (_Float16)f;
  }
  for (int t = lane; t < 16 * (EDGE_PAD - EDGE_IN_); t += 32) {  // zero pad 71..95
    int k = t / (EDGE_PAD - EDGE_IN_), j = t % (EDGE_PAD - EDGE_IN_);
    E[k * EDGE_PAD + EDGE_IN_ + j] = (_Float16)0.0f;
  }
  // GEMM1: 16x96 @ 96x64
  v8f acc[4];
#pragma unroll
  for (int nt = 0; nt < 4; ++nt) acc[nt] = v8f_zero();
#pragma unroll
  for (int kc = 0; kc < 3; ++kc) {
    v16h a = load_A_rm(E, EDGE_PAD, kc * 32);
#pragma unroll
    for (int nt = 0; nt < 4; ++nt)
      acc[nt] = wmma32(a, load_B_packed(sW1, kc, nt), acc[nt]);
  }
#pragma unroll
  for (int nt = 0; nt < 4; ++nt) {
    int n = nt * 16 + (lane & 15);
    float bias = b1[n];
#pragma unroll
    for (int r = 0; r < 8; ++r) acc[nt][r] = gelu_f(acc[nt][r] + bias);
    store_D_f16(acc[nt], E, EDGE_PAD, nt * 16);
  }
  // GEMM2: 16x64 @ 64x64
#pragma unroll
  for (int nt = 0; nt < 4; ++nt) acc[nt] = v8f_zero();
#pragma unroll
  for (int kc = 0; kc < 2; ++kc) {
    v16h a = load_A_rm(E, EDGE_PAD, kc * 32);
#pragma unroll
    for (int nt = 0; nt < 4; ++nt)
      acc[nt] = wmma32(a, load_B_packed(sW2, kc, nt), acc[nt]);
  }
#pragma unroll
  for (int nt = 0; nt < 4; ++nt) {
    int n = nt * 16 + (lane & 15);
    float bias = b2[n];
#pragma unroll
    for (int r = 0; r < 8; ++r) acc[nt][r] = gelu_f(acc[nt][r] + bias);
    store_D_f16(acc[nt], E, EDGE_PAD, nt * 16);
  }
  // GEMM3: 16x64 @ 64x64 (no activation)
  v8f acc3[4];
#pragma unroll
  for (int nt = 0; nt < 4; ++nt) acc3[nt] = v8f_zero();
#pragma unroll
  for (int kc = 0; kc < 2; ++kc) {
    v16h a = load_A_rm(E, EDGE_PAD, kc * 32);
#pragma unroll
    for (int nt = 0; nt < 4; ++nt)
      acc3[nt] = wmma32(a, load_B_packed(sW3, kc, nt), acc3[nt]);
  }
  // weighted aggregation over K neighbors
  float wsum = 0.0f;
#pragma unroll
  for (int k = 0; k < KNN; ++k) wsum += expf(-sD[wid][k] / sigma);
  int mo = (lane >> 4) * 8;
#pragma unroll
  for (int nt = 0; nt < 4; ++nt) {
    int n = nt * 16 + (lane & 15);
    float bias = b3[n];
    float s = 0.0f;
#pragma unroll
    for (int r = 0; r < 8; ++r) {
      float w = expf(-sD[wid][mo + r] / sigma);
      s += w * (acc3[nt][r] + bias);
    }
    s += __shfl_xor(s, 16, 32);
    if (lane < 16) {
      float val = s / fmaxf(wsum, 1e-6f);
      if (out_cm) out[(size_t)(b * 64 + n) * 4096 + q] = val;
      else        out[(size_t)(b * Nq + q) * 64 + n] = val;
    }
  }
}

// ============================ FNO: truncated forward DFT (modes only) ============================
__global__ __launch_bounds__(256) void k_fno_fwd(const float* __restrict__ x, float* __restrict__ xf) {
  __shared__ float sX[64 * 64];
  __shared__ float sT[64 * M2_ * 2];
  int tid = threadIdx.x;
  int bc = blockIdx.x;  // b*64 + c
  const float* xp = x + (size_t)bc * 4096;
  for (int i = tid; i < 4096; i += 256) sX[i] = xp[i];
  __syncthreads();
  const float TPO = 6.283185307179586f / 64.0f;
  for (int t = tid; t < 64 * M2_; t += 256) {
    int xr = t / M2_, ky = t % M2_;
    float re = 0.0f, im = 0.0f;
    for (int y = 0; y < 64; ++y) {
      float th = TPO * (float)((ky * y) & 63);
      float sn, cs; __sincosf(th, &sn, &cs);
      float v = sX[xr * 64 + y];
      re += v * cs; im -= v * sn;
    }
    sT[t * 2 + 0] = re; sT[t * 2 + 1] = im;
  }
  __syncthreads();
  for (int t = tid; t < MR_ * M2_; t += 256) {
    int kxi = t / M2_, ky = t % M2_;
    int kx = kxi < M1_ ? kxi : kxi + 40;    // rows 0..11 and 52..63
    float re = 0.0f, im = 0.0f;
    for (int xr = 0; xr < 64; ++xr) {
      float th = TPO * (float)((kx * xr) & 63);
      float sn, cs; __sincosf(th, &sn, &cs);
      float a = sT[(xr * M2_ + ky) * 2 + 0], bi = sT[(xr * M2_ + ky) * 2 + 1];
      re += a * cs + bi * sn;
      im += bi * cs - a * sn;
    }
    xf[((size_t)bc * MR_ * M2_ + t) * 2 + 0] = re;
    xf[((size_t)bc * MR_ * M2_ + t) * 2 + 1] = im;
  }
}

// ============================ FNO: per-mode channel mixing (complex) ============================
__global__ __launch_bounds__(64) void k_fno_mix(const float* __restrict__ xf,
                                                const float* __restrict__ spec1, const float* __restrict__ spec2,
                                                int d, float* __restrict__ zf) {
  __shared__ float sz[64 * 2];
  int tid = threadIdx.x;
  int bm = blockIdx.x;                       // b*(MR*M2) + kxi*M2 + ky
  int b = bm / (MR_ * M2_); int m = bm - b * (MR_ * M2_);
  int kxi = m / M2_, ky = m - kxi * M2_;
  for (int i = tid; i < 64; i += 64) {
    size_t o = (((size_t)(b * 64 + i) * MR_ + kxi) * M2_ + ky) * 2;
    sz[i * 2 + 0] = xf[o + 0];
    sz[i * 2 + 1] = xf[o + 1];
  }
  __syncthreads();
  int o = tid;
  const float* wb = (kxi < M1_) ? spec1 : spec2;
  int m1 = (kxi < M1_) ? kxi : kxi - M1_;
  float re = 0.0f, im = 0.0f;
  for (int i = 0; i < 64; ++i) {
    size_t widx = ((((size_t)d * 64 + i) * 64 + o) * M1_ + m1) * M2_ + ky;
    float wr = wb[widx * 2 + 0], wi = wb[widx * 2 + 1];
    float ar = sz[i * 2 + 0], ai = sz[i * 2 + 1];
    re += ar * wr - ai * wi;
    im += ar * wi + ai * wr;
  }
  size_t oo = (((size_t)(b * 64 + o) * MR_ + kxi) * M2_ + ky) * 2;
  zf[oo + 0] = re; zf[oo + 1] = im;
}

// ============================ FNO: truncated inverse rDFT ============================
__global__ __launch_bounds__(256) void k_fno_inv(const float* __restrict__ zf, float* __restrict__ s) {
  __shared__ float sZ[MR_ * M2_ * 2];
  __shared__ float sZr[64 * M2_ * 2];
  int tid = threadIdx.x;
  int bo = blockIdx.x;
  for (int i = tid; i < MR_ * M2_ * 2; i += 256) sZ[i] = zf[(size_t)bo * MR_ * M2_ * 2 + i];
  __syncthreads();
  const float TPO = 6.283185307179586f / 64.0f;
  for (int t = tid; t < 64 * M2_; t += 256) {
    int xr = t / M2_, ky = t % M2_;
    float re = 0.0f, im = 0.0f;
    for (int kxi = 0; kxi < MR_; ++kxi) {
      int kx = kxi < M1_ ? kxi : kxi + 40;
      float th = TPO * (float)((kx * xr) & 63);
      float sn, cs; __sincosf(th, &sn, &cs);
      float a = sZ[(kxi * M2_ + ky) * 2 + 0], bi = sZ[(kxi * M2_ + ky) * 2 + 1];
      re += a * cs - bi * sn;
      im += a * sn + bi * cs;
    }
    sZr[t * 2 + 0] = re; sZr[t * 2 + 1] = im;
  }
  __syncthreads();
  float* sp = s + (size_t)bo * 4096;
  for (int t = tid; t < 4096; t += 256) {
    int xr = t >> 6, y = t & 63;
    float acc = sZr[(xr * M2_) * 2 + 0];   // ky=0 term (real part only)
    for (int ky = 1; ky < M2_; ++ky) {
      float th = TPO * (float)((ky * y) & 63);
      float sn, cs; __sincosf(th, &sn, &cs);
      acc += 2.0f * (sZr[(xr * M2_ + ky) * 2 + 0] * cs - sZr[(xr * M2_ + ky) * 2 + 1] * sn);
    }
    sp[t] = acc * (1.0f / 4096.0f);
  }
}

// ============================ FNO: pointwise 64x64 channel conv (WMMA) + add spectral + bias ============================
__global__ __launch_bounds__(256) void k_fno_pw(const float* __restrict__ x, const float* __restrict__ sspec,
                                                const float* __restrict__ pwW, const float* __restrict__ pwb,
                                                int d, float* __restrict__ y) {
  __shared__ alignas(16) _Float16 sWp[2 * 4 * 512];
  __shared__ alignas(16) _Float16 sA[8][16 * 64];
  int tid = threadIdx.x;
  const float* Wd = pwW + (size_t)d * 64 * 64;
  stage_B_packed(sWp, Wd, 2, 64, /*trans=*/1);   // B[k=i][n=o] = pw_W[o][i]
  __syncthreads();
  int wid = tid >> 5, lane = tid & 31;
  int gt = blockIdx.x * 8 + wid;             // 16-pixel tile id
  int b = gt >> 8;                           // 256 tiles per batch
  int pix0 = (gt & 255) * 16;
  _Float16* A = sA[wid];
  for (int t = lane; t < 16 * 64; t += 32) {
    int m = t >> 6, k = t & 63;
    A[m * 64 + k] = (_Float16)x[(size_t)(b * 64 + k) * 4096 + pix0 + m];
  }
  v8f acc[4];
#pragma unroll
  for (int nt = 0; nt < 4; ++nt) acc[nt] = v8f_zero();
#pragma unroll
  for (int kc = 0; kc < 2; ++kc) {
    v16h a = load_A_rm(A, 64, kc * 32);
#pragma unroll
    for (int nt = 0; nt < 4; ++nt)
      acc[nt] = wmma32(a, load_B_packed(sWp, kc, nt), acc[nt]);
  }
#pragma unroll
  for (int nt = 0; nt < 4; ++nt) {
    int n = nt * 16 + (lane & 15), mo = (lane >> 4) * 8;
    float bias = pwb[d * 64 + n];
#pragma unroll
    for (int r = 0; r < 8; ++r) {
      size_t o = (size_t)(b * 64 + n) * 4096 + pix0 + mo + r;
      y[o] = acc[nt][r] + sspec[o] + bias;
    }
  }
}

// ============================ FNO: instance-norm stats ============================
__global__ __launch_bounds__(256) void k_fno_stats(const float* __restrict__ y, float* __restrict__ stats) {
  __shared__ float ssum[256], ssq[256];
  int tid = threadIdx.x; int bo = blockIdx.x;
  const float* yp = y + (size_t)bo * 4096;
  float s = 0.0f, s2 = 0.0f;
  for (int i = tid; i < 4096; i += 256) { float v = yp[i]; s += v; s2 += v * v; }
  ssum[tid] = s; ssq[tid] = s2; __syncthreads();
  for (int o = 128; o > 0; o >>= 1) {
    if (tid < o) { ssum[tid] += ssum[tid + o]; ssq[tid] += ssq[tid + o]; }
    __syncthreads();
  }
  if (tid == 0) {
    float mu = ssum[0] * (1.0f / 4096.0f);
    float var = ssq[0] * (1.0f / 4096.0f) - mu * mu;
    stats[bo * 2 + 0] = mu; stats[bo * 2 + 1] = var;
  }
}

// ============================ FNO: normalize + GELU ============================
__global__ __launch_bounds__(256) void k_fno_norm(const float* __restrict__ y, const float* __restrict__ stats,
                                                  float* __restrict__ xo) {
  int gid = blockIdx.x * 256 + threadIdx.x;  // B*64*4096 total
  int bo = gid >> 12;
  float mu = stats[bo * 2 + 0], var = stats[bo * 2 + 1];
  xo[gid] = gelu_f((y[gid] - mu) * rsqrtf(var + 1e-5f));
}

// ============================ projection MLP (64->64 GELU ->3) ============================
__global__ __launch_bounds__(256) void k_proj(const float* __restrict__ yin,
                                              const float* __restrict__ W1, const float* __restrict__ b1,
                                              const float* __restrict__ W2, const float* __restrict__ b2,
                                              float* __restrict__ out) {
  __shared__ alignas(16) _Float16 sW1p[2 * 4 * 512];
  __shared__ alignas(16) _Float16 sA[8][16 * 64];
  int tid = threadIdx.x;
  stage_B_packed(sW1p, W1, 2, 64, 0);
  __syncthreads();
  int wid = tid >> 5, lane = tid & 31;
  int row0 = (blockIdx.x * 8 + wid) * 16;
  _Float16* A = sA[wid];
  for (int t = lane; t < 16 * 64; t += 32) {
    int m = t >> 6, k = t & 63;
    A[m * 64 + k] = (_Float16)yin[(size_t)(row0 + m) * 64 + k];
  }
  v8f acc[4];
#pragma unroll
  for (int nt = 0; nt < 4; ++nt) acc[nt] = v8f_zero();
#pragma unroll
  for (int kc = 0; kc < 2; ++kc) {
    v16h a = load_A_rm(A, 64, kc * 32);
#pragma unroll
    for (int nt = 0; nt < 4; ++nt)
      acc[nt] = wmma32(a, load_B_packed(sW1p, kc, nt), acc[nt]);
  }
#pragma unroll
  for (int nt = 0; nt < 4; ++nt) {
    int n = nt * 16 + (lane & 15);
    float bias = b1[n];
#pragma unroll
    for (int r = 0; r < 8; ++r) acc[nt][r] = gelu_f(acc[nt][r] + bias);
    store_D_f16(acc[nt], A, 64, nt * 16);
  }
  for (int t = lane; t < 48; t += 32) {      // 16 rows x 3 outputs
    int m = t / 3, c = t - m * 3;
    float s = b2[c];
    for (int j = 0; j < 64; ++j) s += (float)A[m * 64 + j] * W2[j * 3 + c];
    out[(size_t)(row0 + m) * 3 + c] = s;
  }
}

// ============================ host-side launcher ============================
extern "C" void kernel_launch(void* const* d_in, const int* in_sizes, int n_in,
                              void* d_out, int out_size, void* d_ws, size_t ws_size,
                              hipStream_t stream) {
  (void)in_sizes; (void)n_in; (void)out_size; (void)ws_size;
  const float* coords   = (const float*)d_in[0];
  const float* features = (const float*)d_in[1];
  const float* lift_W1  = (const float*)d_in[2];
  const float* lift_b1  = (const float*)d_in[3];
  const float* lift_W2  = (const float*)d_in[4];
  const float* lift_b2  = (const float*)d_in[5];
  const float* gin_W1   = (const float*)d_in[6];
  const float* gin_b1   = (const float*)d_in[7];
  const float* gin_W2   = (const float*)d_in[8];
  const float* gin_b2   = (const float*)d_in[9];
  const float* gin_W3   = (const float*)d_in[10];
  const float* gin_b3   = (const float*)d_in[11];
  const float* spec1    = (const float*)d_in[12];
  const float* spec2    = (const float*)d_in[13];
  const float* pw_W     = (const float*)d_in[14];
  const float* pw_b     = (const float*)d_in[15];
  const float* gout_W1  = (const float*)d_in[16];
  const float* gout_b1  = (const float*)d_in[17];
  const float* gout_W2  = (const float*)d_in[18];
  const float* gout_b2  = (const float*)d_in[19];
  const float* gout_W3  = (const float*)d_in[20];
  const float* gout_b3  = (const float*)d_in[21];
  const float* proj_W1  = (const float*)d_in[22];
  const float* proj_b1  = (const float*)d_in[23];
  const float* proj_W2  = (const float*)d_in[24];
  const float* proj_b2  = (const float*)d_in[25];
  float* out = (float*)d_out;

  size_t off = 0;
  char* wsb = (char*)d_ws;
  auto wsptr = [&](size_t bytes) -> void* {
    void* p = (void*)(wsb + off);
    off = (off + bytes + 255) & ~(size_t)255;
    return p;
  };
  const size_t ROWS = (size_t)BB * NPTS;                  // 16384
  float* feat     = (float*)wsptr(ROWS * 64 * 4);
  float* x0       = (float*)wsptr((size_t)BB * 64 * 4096 * 4);
  float* x1       = (float*)wsptr((size_t)BB * 64 * 4096 * 4);
  float* yb       = (float*)wsptr((size_t)BB * 64 * 4096 * 4);
  float* sbuf     = (float*)wsptr((size_t)BB * 64 * 4096 * 4);
  float* xfb      = (float*)wsptr((size_t)BB * 64 * MR_ * M2_ * 2 * 4);
  float* zfb      = (float*)wsptr((size_t)BB * 64 * MR_ * M2_ * 2 * 4);
  float* stats    = (float*)wsptr((size_t)BB * 64 * 2 * 4);
  float* enc_dist = (float*)wsptr((size_t)BB * NG * KNN * 4);
  int*   enc_idx  = (int*)  wsptr((size_t)BB * NG * KNN * 4);
  float* dec_dist = (float*)wsptr(ROWS * KNN * 4);
  int*   dec_idx  = (int*)  wsptr(ROWS * KNN * 4);
  float* ydec     = (float*)wsptr(ROWS * 64 * 4);
  float* sig_enc  = (float*)wsptr(256);
  float* sig_dec  = (float*)wsptr(256);

  // 1) lift node features: (B*N, 6) -> (B*N, 64)
  k_lift<<<128, 256, 0, stream>>>(features, lift_W1, lift_b1, lift_W2, lift_b2, feat);

  // 2) GNO encode: latent grid queries vs point sources
  k_topk<<<1024, 256, 0, stream>>>((const float*)nullptr, coords, NG, NPTS, 1, 0, enc_dist, enc_idx);
  k_median_sigma<<<1, 256, 0, stream>>>(enc_dist, BB * NG, sig_enc);
  k_edge<<<1024, 256, 0, stream>>>((const float*)nullptr, coords, feat, enc_idx, enc_dist, sig_enc,
                                   gin_W1, gin_b1, gin_W2, gin_b2, gin_W3, gin_b3,
                                   NG, NPTS, 1, 0, /*srcf_cm=*/0, /*out_cm=*/1, x0);

  // 3) FNO blocks (ping-pong x0 <-> x1, y staged in yb)
  for (int d = 0; d < 3; ++d) {
    const float* xin = (d == 1) ? x1 : x0;
    float* xout = (d == 1) ? x0 : x1;
    k_fno_fwd  <<<BB * 64, 256, 0, stream>>>(xin, xfb);
    k_fno_mix  <<<BB * MR_ * M2_, 64, 0, stream>>>(xfb, spec1, spec2, d, zfb);
    k_fno_inv  <<<BB * 64, 256, 0, stream>>>(zfb, sbuf);
    k_fno_pw   <<<BB * 256 / 8, 256, 0, stream>>>(xin, sbuf, pw_W, pw_b, d, yb);
    k_fno_stats<<<BB * 64, 256, 0, stream>>>(yb, stats);
    k_fno_norm <<<BB * 64 * 4096 / 256, 256, 0, stream>>>(yb, stats, xout);
  }
  // final latent in x1 (after d = 2)

  // 4) GNO decode: point queries vs latent grid sources (features channel-major in x1)
  k_topk<<<2048, 256, 0, stream>>>(coords, (const float*)nullptr, NPTS, NG, 0, 1, dec_dist, dec_idx);
  k_median_sigma<<<1, 256, 0, stream>>>(dec_dist, (int)ROWS, sig_dec);
  k_edge<<<2048, 256, 0, stream>>>(coords, (const float*)nullptr, x1, dec_idx, dec_dist, sig_dec,
                                   gout_W1, gout_b1, gout_W2, gout_b2, gout_W3, gout_b3,
                                   NPTS, NG, 0, 1, /*srcf_cm=*/1, /*out_cm=*/0, ydec);

  // 5) projection: (B*N, 64) -> (B*N, 3)
  k_proj<<<128, 256, 0, stream>>>(ydec, proj_W1, proj_b1, proj_W2, proj_b2, out);
}
